// GlobalAttention_38792144617607
// MI455X (gfx1250) — matrix operand order repaired and checked by reading, not compiled, not run on hardware
//
#include <hip/hip_runtime.h>

// GlobalAttention fused kernel set for MI455X (gfx1250), bf16 WMMA + f32 accum.
// B=64, N=2048, D=256, H=8, KD=VD=32, O=256.

#define NB   64
#define NN   2048
#define ND   256
#define NH   8
#define NKD  32
#define NVD  32
#define NO   256
#define LARGEF 1e9f
#define EPSF   1e-10f

typedef __attribute__((ext_vector_type(16))) __bf16 v16bf;
typedef __attribute__((ext_vector_type(8)))  float  v8f;

__device__ __forceinline__ unsigned short f2bf(float x) {
  unsigned u = __float_as_uint(x);
  u += 0x7FFFu + ((u >> 16) & 1u);   // round-to-nearest-even
  return (unsigned short)(u >> 16);
}
__device__ __forceinline__ unsigned pack2(float a, float b) {
  return (unsigned)f2bf(a) | ((unsigned)f2bf(b) << 16);
}

// A-matrix fragment (16x32 bf16, row-major source, stride `s` elements):
// lane half = lane>>4, row = lane&15; K groups [k+8h .. +7] and [k+16+8h .. +7].
__device__ __forceinline__ v16bf load_a_frag(const unsigned short* base, int row,
                                             int half, int kbase, int s) {
  const unsigned short* p = base + row * s + kbase + half * 8;
  union { struct { uint4 lo, hi; } u; v16bf v; } f;
  f.u.lo = *(const uint4*)(p);
  f.u.hi = *(const uint4*)(p + 16);
  return f.v;
}
// B-matrix fragment from K-major (transposed) weights: lane holds column col,
// K groups [k+8h .. +7] and [k+16+8h .. +7].
__device__ __forceinline__ v16bf load_b_frag(const unsigned short* base, int col,
                                             int half, int kbase, int s) {
  const unsigned short* p = base + col * s + kbase + half * 8;
  union { struct { uint4 lo, hi; } u; v16bf v; } f;
  f.u.lo = *(const uint4*)(p);
  f.u.hi = *(const uint4*)(p + 16);
  return f.v;
}

// ---------------------------------------------------------------- kernel 1 --
// Convert weights to bf16, K-major layout for B-fragments.
__global__ void wconv_kernel(const float* __restrict__ gating_w,
                             const float* __restrict__ output_w,
                             const float* __restrict__ key_w,
                             const float* __restrict__ value_w,
                             unsigned short* __restrict__ wg_t,
                             unsigned short* __restrict__ wo_t,
                             unsigned short* __restrict__ wkv_t) {
  int idx = blockIdx.x * 256 + threadIdx.x;
  if (idx < 65536) {                       // Wg_t[hv][d] <- gating_w[d][hv]
    int c = idx >> 8, d = idx & 255;
    wg_t[idx] = f2bf(gating_w[d * 256 + c]);
  } else if (idx < 131072) {               // Wo_t[o][hv] <- output_w[hv][o]
    int j = idx - 65536;
    int o = j >> 8, hv = j & 255;
    wo_t[j] = f2bf(output_w[hv * 256 + o]);
  } else if (idx < 147456) {               // Wkv_t[c][d]: c<32 key, else value
    int j = idx - 131072;
    int c = j >> 8, d = j & 255;
    wkv_t[j] = f2bf(c < 32 ? key_w[d * 32 + c] : value_w[d * 32 + (c - 32)]);
  }
}

// ---------------------------------------------------------------- kernel 2 --
// q_avg (masked mean over N) then q[b,h,kd] = q_avg . query_w * KD^-0.5
__global__ void qavg_kernel(const float* __restrict__ q_data,
                            const float* __restrict__ q_mask,
                            const float* __restrict__ query_w,
                            float* __restrict__ q_out) {
  int b = blockIdx.x, t = threadIdx.x;
  __shared__ float qa[ND];
  const float* qd = q_data + (size_t)b * NN * ND + t;
  const float* mk = q_mask + (size_t)b * NN;
  float s = 0.f, ms = 0.f;
  for (int n = 0; n < NN; ++n) {
    float m = mk[n];
    s += m * qd[(size_t)n * ND];
    ms += m;
  }
  qa[t] = s / (ms + EPSF);
  __syncthreads();
  float acc = 0.f;
  for (int d = 0; d < ND; ++d) acc += qa[d] * query_w[d * 256 + t];  // t = h*32+kd
  q_out[b * 256 + t] = acc * 0.17677669529663687f;                   // 1/sqrt(32)
}

// ---------------------------------------------------------------- kernel 3 --
// Per 16-row tile of m_data: KV = X * Wkv (WMMA), stash v (f32), logits = q.k + bias.
__global__ void kv_logits_kernel(const float* __restrict__ m_data,
                                 const float* __restrict__ q_mask,
                                 const unsigned short* __restrict__ wkv_t,
                                 const float* __restrict__ q_ws,
                                 float* __restrict__ vbuf,
                                 float* __restrict__ logits) {
  int tile = blockIdx.x, b = blockIdx.y;
  int t = threadIdx.x;
  int wave = t >> 5, lane = t & 31;
  int half = lane >> 4, idx = lane & 15;

  __shared__ unsigned short Xs[16 * ND];   // m_data tile, bf16
  __shared__ float Ks[16 * NKD];           // k tile, f32
  __shared__ float Qs[256];                // q[b,:,:]

  {  // stage 16x256 tile as bf16: 128 threads x 32 floats
    int row = t >> 3, chunk = t & 7;
    const float* src = m_data + ((size_t)(b * NN + tile * 16 + row) * ND + chunk * 32);
    unsigned* dst = (unsigned*)&Xs[row * ND + chunk * 32];
#pragma unroll
    for (int i = 0; i < 8; ++i) {
      float4 f = ((const float4*)src)[i];
      dst[2 * i]     = pack2(f.x, f.y);
      dst[2 * i + 1] = pack2(f.z, f.w);
    }
  }
  Qs[t] = q_ws[b * 256 + t];
  Qs[t + 128] = q_ws[b * 256 + t + 128];
  __syncthreads();

  if (wave < 2) {  // wave0: k cols 0..31, wave1: v cols 32..63
    v8f acc0 = {};
    v8f acc1 = {};
    int c0 = wave * 32;
#pragma unroll
    for (int k = 0; k < ND; k += 32) {
      v16bf a  = load_a_frag(Xs, idx, half, k, ND);
      v16bf b0 = load_b_frag(wkv_t, c0 + idx, half, k, ND);
      v16bf b1 = load_b_frag(wkv_t, c0 + 16 + idx, half, k, ND);
      acc0 = __builtin_amdgcn_wmma_f32_16x16x32_bf16(false, a, false, b0, (short)0, acc0, false, false);
      acc1 = __builtin_amdgcn_wmma_f32_16x16x32_bf16(false, a, false, b1, (short)0, acc1, false, false);
    }
    if (wave == 0) {
#pragma unroll
      for (int r = 0; r < 8; ++r) {
        int m = half * 8 + r;
        Ks[m * NKD + idx]      = acc0[r];
        Ks[m * NKD + 16 + idx] = acc1[r];
      }
    } else {
      float* vb = vbuf + (size_t)(b * NN + tile * 16) * NVD;
#pragma unroll
      for (int r = 0; r < 8; ++r) {
        int m = half * 8 + r;
        vb[m * NVD + idx]      = acc0[r];
        vb[m * NVD + 16 + idx] = acc1[r];
      }
    }
  }
  __syncthreads();

  {  // logits: 128 threads = (h, m)
    int h = t >> 4, m = t & 15;
    float s = 0.f;
#pragma unroll
    for (int kd = 0; kd < NKD; ++kd) s += Qs[h * 32 + kd] * Ks[m * NKD + kd];
    float mv = q_mask[(size_t)b * NN + tile * 16 + m];
    logits[((size_t)(b * NH) + h) * NN + tile * 16 + m] = s + LARGEF * (mv - 1.f);
  }
}

// ---------------------------------------------------------------- kernel 4 --
// Softmax over N and weighted_avg[b,h,v] = sum_n w[n] * v[b,n,v].
__global__ void softmax_wavg_kernel(const float* __restrict__ logits,
                                    const float* __restrict__ vbuf,
                                    float* __restrict__ wavg) {
  int bh = blockIdx.x;
  int b = bh >> 3, h = bh & 7;
  int t = threadIdx.x;
  const float* lg = logits + (size_t)bh * NN;
  __shared__ float red[256];

  float mx = -3.4e38f;
  for (int n = t; n < NN; n += 256) mx = fmaxf(mx, lg[n]);
  red[t] = mx;
  __syncthreads();
  for (int s = 128; s > 0; s >>= 1) {
    if (t < s) red[t] = fmaxf(red[t], red[t + s]);
    __syncthreads();
  }
  mx = red[0];
  __syncthreads();

  float se = 0.f;
  for (int n = t; n < NN; n += 256) se += __expf(lg[n] - mx);
  red[t] = se;
  __syncthreads();
  for (int s = 128; s > 0; s >>= 1) {
    if (t < s) red[t] += red[t + s];
    __syncthreads();
  }
  float Z = red[0];
  __syncthreads();

  int v = t & 31, g = t >> 5;
  float acc = 0.f;
  const float* vb = vbuf + (size_t)b * NN * NVD;
  for (int n = g; n < NN; n += 8) acc += __expf(lg[n] - mx) * vb[(size_t)n * NVD + v];
  red[t] = acc;
  __syncthreads();
  if (t < 32) {
    float s2 = 0.f;
#pragma unroll
    for (int j = 0; j < 8; ++j) s2 += red[t + 32 * j];
    wavg[b * 256 + h * 32 + t] = s2 / Z;
  }
}

// ---------------------------------------------------------------- kernel 5 --
// Fused: gate = sigmoid(Xq*Wg + b); G' = wavg .* gate; out = G'*Wo + output_b.
// One block = 64 query rows (4 M-tiles), 8 waves; each wave owns two 16-col
// tiles across all 4 M-tiles (8 accumulators), so each B-fragment load feeds
// 4 WMMAs -> 4x less weight/L2 traffic than 16-row blocking.
__global__ void fused_gate_out_kernel(const float* __restrict__ q_data,
                                      const unsigned short* __restrict__ wg_t,
                                      const unsigned short* __restrict__ wo_t,
                                      const float* __restrict__ gating_b,
                                      const float* __restrict__ wavg,
                                      const float* __restrict__ output_b,
                                      float* __restrict__ out) {
  int tile = blockIdx.x, b = blockIdx.y;       // tile of 64 rows
  int row0 = tile * 64;
  int t = threadIdx.x;
  int wave = t >> 5, lane = t & 31;
  int half = lane >> 4, idx = lane & 15;

  __shared__ unsigned short Xs[64 * ND];   // q_data tile bf16 (32KB)
  __shared__ unsigned short Gs[64 * ND];   // gated tile bf16 (32KB)

  {  // stage 64x256 tile: 256 threads x 64 floats
    int row = t >> 2, chunk = t & 3;
    const float* src = q_data + ((size_t)(b * NN + row0 + row) * ND + chunk * 64);
    unsigned* dst = (unsigned*)&Xs[row * ND + chunk * 64];
#pragma unroll
    for (int i = 0; i < 16; ++i) {
      float4 f = ((const float4*)src)[i];
      dst[2 * i]     = pack2(f.x, f.y);
      dst[2 * i + 1] = pack2(f.z, f.w);
    }
  }
  __syncthreads();

  int n0 = wave * 32;
  int na = n0 + idx, nb = n0 + 16 + idx;

  // GEMM1: gate logits, 4 M-tiles x 2 N-tiles per wave
  v8f acc[4][2];
#pragma unroll
  for (int mt = 0; mt < 4; ++mt) {
    acc[mt][0] = (v8f){};
    acc[mt][1] = (v8f){};
  }
#pragma unroll
  for (int k = 0; k < ND; k += 32) {
    v16bf b0 = load_b_frag(wg_t, na, half, k, ND);
    v16bf b1 = load_b_frag(wg_t, nb, half, k, ND);
#pragma unroll
    for (int mt = 0; mt < 4; ++mt) {
      v16bf a = load_a_frag(Xs, mt * 16 + idx, half, k, ND);
      acc[mt][0] = __builtin_amdgcn_wmma_f32_16x16x32_bf16(false, a, false, b0, (short)0, acc[mt][0], false, false);
      acc[mt][1] = __builtin_amdgcn_wmma_f32_16x16x32_bf16(false, a, false, b1, (short)0, acc[mt][1], false, false);
    }
  }
  // epilogue: sigmoid + fold weighted_avg, write bf16 A-tile for GEMM2
  {
    float ba = gating_b[na], bb = gating_b[nb];
    float wa = wavg[b * 256 + na], wb = wavg[b * 256 + nb];
#pragma unroll
    for (int mt = 0; mt < 4; ++mt) {
#pragma unroll
      for (int r = 0; r < 8; ++r) {
        int m = mt * 16 + half * 8 + r;
        float g0 = wa / (1.f + __expf(-(acc[mt][0][r] + ba)));
        float g1 = wb / (1.f + __expf(-(acc[mt][1][r] + bb)));
        Gs[m * ND + na] = f2bf(g0);
        Gs[m * ND + nb] = f2bf(g1);
      }
    }
  }
  __syncthreads();

  // GEMM2: output projection
#pragma unroll
  for (int mt = 0; mt < 4; ++mt) {
    acc[mt][0] = (v8f){};
    acc[mt][1] = (v8f){};
  }
#pragma unroll
  for (int k = 0; k < ND; k += 32) {
    v16bf b0 = load_b_frag(wo_t, na, half, k, ND);
    v16bf b1 = load_b_frag(wo_t, nb, half, k, ND);
#pragma unroll
    for (int mt = 0; mt < 4; ++mt) {
      v16bf a = load_a_frag(Gs, mt * 16 + idx, half, k, ND);
      acc[mt][0] = __builtin_amdgcn_wmma_f32_16x16x32_bf16(false, a, false, b0, (short)0, acc[mt][0], false, false);
      acc[mt][1] = __builtin_amdgcn_wmma_f32_16x16x32_bf16(false, a, false, b1, (short)0, acc[mt][1], false, false);
    }
  }
  float ob0 = output_b[na], ob1 = output_b[nb];
  float* dst = out + (size_t)(b * NN + row0) * NO;
#pragma unroll
  for (int mt = 0; mt < 4; ++mt) {
#pragma unroll
    for (int r = 0; r < 8; ++r) {
      int m = mt * 16 + half * 8 + r;
      dst[(size_t)m * NO + na] = acc[mt][0][r] + ob0;
      dst[(size_t)m * NO + nb] = acc[mt][1][r] + ob1;
    }
  }
}

// ------------------------------------------------------------------ launch --
extern "C" void kernel_launch(void* const* d_in, const int* in_sizes, int n_in,
                              void* d_out, int out_size, void* d_ws, size_t ws_size,
                              hipStream_t stream) {
  (void)in_sizes; (void)n_in; (void)out_size; (void)ws_size;
  const float* q_data   = (const float*)d_in[0];
  const float* m_data   = (const float*)d_in[1];
  const float* q_mask   = (const float*)d_in[2];
  const float* query_w  = (const float*)d_in[3];
  const float* key_w    = (const float*)d_in[4];
  const float* value_w  = (const float*)d_in[5];
  const float* gating_w = (const float*)d_in[6];
  const float* gating_b = (const float*)d_in[7];
  const float* output_w = (const float*)d_in[8];
  const float* output_b = (const float*)d_in[9];
  float* out = (float*)d_out;

  char* ws = (char*)d_ws;
  size_t off = 0;
  float* q_ws = (float*)(ws + off);            off += (size_t)NB * 256 * 4;        // 64 KB
  float* wavg = (float*)(ws + off);            off += (size_t)NB * 256 * 4;        // 64 KB
  float* logits = (float*)(ws + off);          off += (size_t)NB * NH * NN * 4;    // 4 MB
  unsigned short* wg_t = (unsigned short*)(ws + off);  off += (size_t)256 * 256 * 2;
  unsigned short* wo_t = (unsigned short*)(ws + off);  off += (size_t)256 * 256 * 2;
  unsigned short* wkv_t = (unsigned short*)(ws + off); off += (size_t)64 * 256 * 2;
  float* vbuf = (float*)(ws + off);            off += (size_t)NB * NN * NVD * 4;   // 16 MB

  wconv_kernel<<<576, 256, 0, stream>>>(gating_w, output_w, key_w, value_w,
                                        wg_t, wo_t, wkv_t);
  qavg_kernel<<<NB, 256, 0, stream>>>(q_data, q_mask, query_w, q_ws);
  kv_logits_kernel<<<dim3(NN / 16, NB), 128, 0, stream>>>(m_data, q_mask, wkv_t,
                                                          q_ws, vbuf, logits);
  softmax_wavg_kernel<<<NB * NH, 256, 0, stream>>>(logits, vbuf, wavg);
  fused_gate_out_kernel<<<dim3(NN / 64, NB), 256, 0, stream>>>(q_data, wg_t, wo_t,
                                                               gating_b, wavg,
                                                               output_b, out);
}
